// LightGRULayer_85426899517830
// MI455X (gfx1250) — compile-verified, matrix operand order checked by reading, not run on hardware
//
#include <hip/hip_runtime.h>

#define SEQ   2048
#define BATCH 64
#define DIN   256
#define DHID  256
#define MBLK  128   // xproj rows per block

typedef __attribute__((ext_vector_type(16))) _Float16 v16h;
typedef __attribute__((ext_vector_type(8)))  _Float16 v8h;
typedef __attribute__((ext_vector_type(8)))  float    v8f;
typedef __attribute__((ext_vector_type(4)))  int      v4i;

static __device__ __forceinline__ v8f wmma_f16(v16h a, v16h b, v8f c) {
    // D = A(16x32 f16) * B(32x16 f16) + C(16x16 f32)
    return __builtin_amdgcn_wmma_f32_16x16x32_f16(
        /*neg_a=*/false, a, /*neg_b=*/false, b,
        /*c_mod=*/(short)0, c, /*reuse_a=*/false, /*reuse_b=*/false);
}

// --- CDNA5 async global->LDS path (guarded: falls back to sync copy) -------
#if defined(__AMDGCN__) && \
    __has_builtin(__builtin_amdgcn_global_load_async_to_lds_b128) && \
    __has_builtin(__builtin_amdgcn_s_wait_asynccnt)
#define HAVE_ASYNC_LDS 1
#endif

static __device__ __forceinline__ void lds_copy_b128(const _Float16* g, _Float16* l) {
#ifdef HAVE_ASYNC_LDS
    __builtin_amdgcn_global_load_async_to_lds_b128(
        (__attribute__((address_space(1))) v4i*)(g),
        (__attribute__((address_space(3))) v4i*)(l), 0, 0);
#else
    *(v8h*)l = *(const v8h*)g;
#endif
}
static __device__ __forceinline__ void lds_copy_fence() {
#ifdef HAVE_ASYNC_LDS
    __builtin_amdgcn_s_wait_asynccnt(0);
#endif
}

// ---------------------------------------------------------------------------
// fp32 -> f16 elementwise convert (x)
// ---------------------------------------------------------------------------
__global__ void k_cvt_f32_to_f16(const float* __restrict__ src,
                                 _Float16* __restrict__ dst, int n) {
    int i = blockIdx.x * blockDim.x + threadIdx.x;
    if (i < n) dst[i] = (_Float16)src[i];
}

// ---------------------------------------------------------------------------
// Pack (N x K) row-major fp32 weight into per-lane-contiguous f16 B fragments:
//   dst[((kt*(N/16)+nt)*32 + lane)*16 + j], n = nt*16+lane%16,
//   k = kt*32 + (lane/16)*16 + j  -> each B fragment is ONE v16h load.
// ---------------------------------------------------------------------------
__global__ void k_pack_b(const float* __restrict__ src,
                         _Float16* __restrict__ dst, int N, int K) {
    int tid = blockIdx.x * blockDim.x + threadIdx.x;
    if (tid >= N * K) return;
    int j    = tid & 15;
    int lane = (tid >> 4) & 31;
    int nt   = (tid >> 9) % (N >> 4);
    int kt   = tid / ((N >> 4) << 9);
    int n = nt * 16 + (lane & 15);
    int k = kt * 32 + ((lane >> 4) << 4) + j;
    dst[tid] = (_Float16)src[n * K + k];
}

// ---------------------------------------------------------------------------
// Input projections: xz = x*Wz^T + bz ; xh = x*Wh^T + bh
// 128-row M-block staged in LDS (async); each wave owns 4 (gate,ntile) combos
// and keeps the 8 B fragments in registers across 8 M-tiles (8x B reuse).
// M-tile loop deliberately NOT unrolled: one acc set + one A set live ->
// no VGPR spills (A re-read from LDS; latency hidden by 8 waves/block).
// Output stored f16 transposed per step: [t][n][b].
// ---------------------------------------------------------------------------
__global__ __launch_bounds__(256) void k_xproj(
    const _Float16* __restrict__ x16,                    // (SEQ*BATCH, DIN)
    const _Float16* __restrict__ WzP, const _Float16* __restrict__ WhP,
    const float* __restrict__ bz, const float* __restrict__ bh,
    _Float16* __restrict__ xzT, _Float16* __restrict__ xhT) // (SEQ, DHID, BATCH)
{
    extern __shared__ _Float16 At[];                     // MBLK*DIN halfs = 64 KB
    const int m0 = blockIdx.x * MBLK;

    // cooperative stage: MBLK*DIN = 32768 halfs; 256 threads x 16 b128 ops
#pragma unroll
    for (int c = 0; c < 16; ++c) {
        int idx = (c * 256 + threadIdx.x) * 8;           // half index
        lds_copy_b128(&x16[(size_t)m0 * DIN + idx], &At[idx]);
    }
    lds_copy_fence();
    __syncthreads();

    const int wave = threadIdx.x >> 5;
    const int lane = threadIdx.x & 31;
    const int koff = (lane >> 4) << 4;
    const int bhalf = (lane >> 4) << 3;

#pragma unroll 1
    for (int combo = 0; combo < 4; ++combo) {
        int jid  = wave * 4 + combo;                     // 32 = 16 ntiles x 2 gates
        int gate = jid >> 4;
        int nt   = jid & 15;
        const _Float16* Bp   = gate ? WhP : WzP;
        const float*    bias = gate ? bh  : bz;
        _Float16*       dst  = gate ? xhT : xzT;

        v16h B[8];                                       // resident across M-tiles
#pragma unroll
        for (int kt = 0; kt < 8; ++kt)
            B[kt] = *(const v16h*)&Bp[(size_t)(((kt * 16 + nt) * 32) + lane) * 16];

        int n = nt * 16 + (lane & 15);
        float bv = bias[n];

#pragma unroll 1
        for (int mt = 0; mt < 8; ++mt) {
            v8f acc = {};
#pragma unroll
            for (int kt = 0; kt < 8; ++kt) {
                v16h a = *(const v16h*)&At[(mt * 16 + (lane & 15)) * DIN + kt * 32 + koff];
                acc = wmma_f16(a, B[kt], acc);
            }
            int m  = m0 + mt * 16;
            int t  = m >> 6;                             // 64 batch rows / step
            int bl = (m & 63) + bhalf;
            v8h pk;
#pragma unroll
            for (int r = 0; r < 8; ++r) pk[r] = (_Float16)(acc[r] + bv);
            *(v8h*)&dst[((size_t)t * DHID + n) * BATCH + bl] = pk;
        }
    }
}

// ---------------------------------------------------------------------------
// Recurrence: 4 independent blocks, each owns 16 batch rows (batch-parallel,
// zero cross-block sync).  Uz/Uh packed in LDS (256 KB), h double-buffered in
// LDS (one barrier per step), h_old carried in registers (C-fragment layout
// is step-invariant), outputs streamed with non-temporal stores.
// ---------------------------------------------------------------------------
__global__ __launch_bounds__(512) void k_recur(
    const float* __restrict__ hidden,
    const _Float16* __restrict__ UzP, const _Float16* __restrict__ UhP,
    const float* __restrict__ cz, const float* __restrict__ ch,
    const _Float16* __restrict__ xzT, const _Float16* __restrict__ xhT,
    float* __restrict__ out)
{
    extern __shared__ _Float16 smem[];
    _Float16* Uzs = smem;                        // 65536 halfs = 128 KB
    _Float16* Uhs = smem + 65536;                // 128 KB
    _Float16* hb0 = smem + 131072;               // 16 x 256 halfs
    _Float16* hb1 = hb0 + 16 * DHID;             // 16 x 256 halfs

    const int tid = threadIdx.x;
    const int b0  = blockIdx.x * 16;

    // Stage packed U matrices into LDS (async when available)
#pragma unroll
    for (int c = 0; c < 16; ++c) {
        int off = (c * 512 + tid) * 8;
        lds_copy_b128(&UzP[off], &Uzs[off]);
        lds_copy_b128(&UhP[off], &Uhs[off]);
    }
    lds_copy_fence();

    const int wave = tid >> 5;                   // 16 waves <-> 16 n-tiles
    const int lane = tid & 31;
    const int nt   = wave;
    const int n    = nt * 16 + (lane & 15);
    const int mrel = (lane >> 4) << 3;           // C-fragment row base (0 or 8)
    const int koff = (lane >> 4) << 4;
    const float czv = cz[n], chv = ch[n];

    // h_old lives in registers: lane (l), vgpr (r) <-> (m = mrel+r, n) matches
    // the C-fragment layout exactly, every step.
    float hnew[8];
#pragma unroll
    for (int r = 0; r < 8; ++r) {
        hnew[r] = hidden[(size_t)(b0 + mrel + r) * DHID + n];
        hb0[(mrel + r) * DHID + n] = (_Float16)hnew[r];  // also fills buffer 0
    }
    __syncthreads();

#pragma unroll 1
    for (int t = 0; t < SEQ; ++t) {
        _Float16* cur = (t & 1) ? hb1 : hb0;
        _Float16* nxt = (t & 1) ? hb0 : hb1;

        // A fragments from current h buffer (shared across both gates)
        v16h a[8];
#pragma unroll
        for (int kt = 0; kt < 8; ++kt)
            a[kt] = *(const v16h*)&cur[(lane & 15) * DHID + kt * 32 + koff];

        v8f az = {}, ah = {};
#pragma unroll
        for (int kt = 0; kt < 8; ++kt) {
            v16h bzf = *(const v16h*)&Uzs[(size_t)(((kt * 16 + nt) * 32) + lane) * 16];
            az = wmma_f16(a[kt], bzf, az);
        }
#pragma unroll
        for (int kt = 0; kt < 8; ++kt) {
            v16h bhf = *(const v16h*)&Uhs[(size_t)(((kt * 16 + nt) * 32) + lane) * 16];
            ah = wmma_f16(a[kt], bhf, ah);
        }

        const size_t xoff = ((size_t)t * DHID + n) * BATCH + b0 + mrel;
        v8h xzv = *(const v8h*)&xzT[xoff];
        v8h xhv = *(const v8h*)&xhT[xoff];
        if (t + 1 < SEQ) {
            __builtin_prefetch(&xzT[xoff + (size_t)DHID * BATCH], 0, 1);
            __builtin_prefetch(&xhT[xoff + (size_t)DHID * BATCH], 0, 1);
        }

#pragma unroll
        for (int r = 0; r < 8; ++r) {
            float hold = hnew[r];                        // register-carried h_old
            float zp = az[r] + (float)xzv[r] + czv;
            float z  = 1.0f / (1.0f + __expf(-zp));
            float hp = ah[r] + (float)xhv[r] + chv;
            float e2 = __expf(-2.0f * hp);
            float th = (1.0f - e2) / (1.0f + e2);
            hnew[r] = hold + z * (th - hold);
        }

        // publish next h (double buffer -> single barrier per step)
#pragma unroll
        for (int r = 0; r < 8; ++r)
            nxt[(mrel + r) * DHID + n] = (_Float16)hnew[r];

        // stream outputs (write-once data: non-temporal, keep L2 for xz/xh)
        float* op = out + ((size_t)t * BATCH + b0 + mrel) * DHID + n;
#pragma unroll
        for (int r = 0; r < 8; ++r)
            __builtin_nontemporal_store(hnew[r], &op[(size_t)r * DHID]);

        __syncthreads();
    }

    // final hidden state appended after the outputs
    float* hf = out + (size_t)SEQ * BATCH * DHID;
#pragma unroll
    for (int r = 0; r < 8; ++r)
        hf[(size_t)(b0 + mrel + r) * DHID + n] = hnew[r];
}

// ---------------------------------------------------------------------------
extern "C" void kernel_launch(void* const* d_in, const int* in_sizes, int n_in,
                              void* d_out, int out_size, void* d_ws, size_t ws_size,
                              hipStream_t stream) {
    (void)in_sizes; (void)n_in; (void)out_size; (void)ws_size;
    const float* x      = (const float*)d_in[0];
    const float* hidden = (const float*)d_in[1];
    const float* Wz     = (const float*)d_in[2];
    const float* bz     = (const float*)d_in[3];
    const float* Uz     = (const float*)d_in[4];
    const float* cz     = (const float*)d_in[5];
    const float* Wh     = (const float*)d_in[6];
    const float* bh     = (const float*)d_in[7];
    const float* Uh     = (const float*)d_in[8];
    const float* ch     = (const float*)d_in[9];
    float* out = (float*)d_out;

    char* ws = (char*)d_ws;
    size_t off = 0;
    auto alloc = [&](size_t bytes) -> void* {
        off = (off + 255) & ~(size_t)255;
        void* p = ws + off;
        off += bytes;
        return p;
    };
    const size_t NX = (size_t)SEQ * BATCH * DIN;
    _Float16* x16 = (_Float16*)alloc(NX * sizeof(_Float16));
    _Float16* WzP = (_Float16*)alloc((size_t)DHID * DIN  * sizeof(_Float16));
    _Float16* WhP = (_Float16*)alloc((size_t)DHID * DIN  * sizeof(_Float16));
    _Float16* UzP = (_Float16*)alloc((size_t)DHID * DHID * sizeof(_Float16));
    _Float16* UhP = (_Float16*)alloc((size_t)DHID * DHID * sizeof(_Float16));
    _Float16* xzT = (_Float16*)alloc((size_t)SEQ * DHID * BATCH * sizeof(_Float16));
    _Float16* xhT = (_Float16*)alloc((size_t)SEQ * DHID * BATCH * sizeof(_Float16));

    // 1) x -> f16
    k_cvt_f32_to_f16<<<(int)((NX + 255) / 256), 256, 0, stream>>>(x, x16, (int)NX);
    // 2) pack weights into WMMA B-fragment order
    k_pack_b<<<(DHID * DIN  + 255) / 256, 256, 0, stream>>>(Wz, WzP, DHID, DIN);
    k_pack_b<<<(DHID * DIN  + 255) / 256, 256, 0, stream>>>(Wh, WhP, DHID, DIN);
    k_pack_b<<<(DHID * DHID + 255) / 256, 256, 0, stream>>>(Uz, UzP, DHID, DHID);
    k_pack_b<<<(DHID * DHID + 255) / 256, 256, 0, stream>>>(Uh, UhP, DHID, DHID);
    // 3) batched input projections (WMMA, 128-row M blocks, B in registers)
    const size_t xp_smem = (size_t)MBLK * DIN * sizeof(_Float16);   // 64 KB
    (void)hipFuncSetAttribute((const void*)k_xproj,
                              hipFuncAttributeMaxDynamicSharedMemorySize, (int)xp_smem);
    k_xproj<<<SEQ * BATCH / MBLK, 256, xp_smem, stream>>>(x16, WzP, WhP, bz, bh,
                                                          xzT, xhT);
    // 4) recurrence: 4 independent batch-slices, LDS-resident weights,
    //    double-buffered h
    const size_t rc_smem = (size_t)(2 * 65536 + 2 * 16 * DHID) * sizeof(_Float16); // 272 KB
    (void)hipFuncSetAttribute((const void*)k_recur,
                              hipFuncAttributeMaxDynamicSharedMemorySize, (int)rc_smem);
    k_recur<<<BATCH / 16, 512, rc_smem, stream>>>(hidden, UzP, UhP, cz, ch,
                                                  xzT, xhT, out);
}
